// Seq2SeqGRU_89197880803727
// MI455X (gfx1250) — matrix operand order repaired and checked by reading, not compiled, not run on hardware
//
#include <hip/hip_runtime.h>

// ---------------------------------------------------------------------------
// Seq2Seq GRU for MI455X (gfx1250, wave32, WMMA bf16 path)
//   encoder: 336 fused GRU-cell steps (one launch each, h ping-pong)
//   decoder: 48 fused GRU-cell steps + linear head (y -> d_out, bf16 feedback)
// GRU cell uses 2xM register blocking: each wave computes a 32x16 slab of
// h_new, so every Whh B-fragment load feeds two v_wmma_f32_16x16x32_bf16 ops
// (halves L2 B traffic; per-step work is L2-BW bound, not WMMA bound).
// ---------------------------------------------------------------------------

typedef __bf16 v16bf __attribute__((ext_vector_type(16)));
typedef __bf16 v8bf  __attribute__((ext_vector_type(8)));
typedef float  v8f   __attribute__((ext_vector_type(8)));

constexpr int SEQ  = 336;
constexpr int BAT  = 512;
constexpr int FEA  = 64;
constexpr int HID  = 512;
constexpr int TOUT = 48;

#define DEVINL __device__ __forceinline__

DEVINL v8f wmma_bf16(v16bf a, v16bf b, v8f c) {
  // (neg_a, A, neg_b, B, c_mod, C, reuse_a, reuse_b)
  return __builtin_amdgcn_wmma_f32_16x16x32_bf16(false, a, false, b,
                                                 (short)0, c, false, false);
}

// Per-lane A fragment for 16-bit 16x32 A: elements [0..15] contiguous
// relative to (row base + k0 + (lane>=16 ? 16 : 0)).
DEVINL v16bf ld_a16(const __bf16* base) {
  v8bf lo = *(const v8bf*)(base);
  v8bf hi = *(const v8bf*)(base + 8);
  return __builtin_shufflevector(lo, hi,
      0, 1, 2, 3, 4, 5, 6, 7, 8, 9, 10, 11, 12, 13, 14, 15);
}

DEVINL float fast_sigmoid(float x) { return 1.0f / (1.0f + __expf(-x)); }
DEVINL float fast_tanh(float x)    { return 1.0f - 2.0f / (__expf(2.0f * x) + 1.0f); }

// ---------------------------------------------------------------------------
// Fused GRU cell, 2xM register blocking.
// Grid: 64 blocks x 256 threads = 512 waves = 16(Mslabs) x 32(Ntiles).
// Each wave: rows [slab*32, slab*32+32), cols [tile_n*16, +16) of [BATxHID].
// ---------------------------------------------------------------------------
__global__ __launch_bounds__(256) void gru_cell_k(
    const __bf16* __restrict__ x_bf,     // [BAT, FEA] bf16
    const __bf16* __restrict__ Wih,      // [3H, FEA] bf16 (rows: r, z, n)
    const __bf16* __restrict__ Whh,      // [3H, HID] bf16
    const float*  __restrict__ bih,      // [3H] fp32
    const float*  __restrict__ bhh,      // [3H] fp32
    const __bf16* __restrict__ h_in_bf,  // [BAT, HID] bf16
    const float*  __restrict__ h_in_f,   // [BAT, HID] fp32
    float*        __restrict__ h_out_f,  // [BAT, HID] fp32
    __bf16*       __restrict__ h_out_bf) // [BAT, HID] bf16
{
  const int wave   = (blockIdx.x * blockDim.x + threadIdx.x) >> 5;
  const int lane   = threadIdx.x & 31;
  const int slab   = wave >> 5;   // 0..15  (32-row slabs of batch)
  const int tile_n = wave & 31;   // 0..31  (16-col tiles of hidden)
  const int lane15 = lane & 15;
  const int lhi    = lane >> 4;

  const int a_row0 = slab * 32 + lane15;       // A rows, first 16x16 M tile
  const int a_row1 = a_row0 + 16;              // second M tile
  const int a_koff = lhi * 16;                 // A per-lane K base
  const int b_koff = lhi * 16;                 // B per-lane K base
  const int n_g    = tile_n * 16 + lane15;     // hidden column (per-lane N)

  v8f r0 = {}, r1 = {}, z0 = {}, z1 = {}, in0 = {}, in1 = {}, hn0 = {}, hn1 = {};

  // ---- gi = x @ Wih^T (K = FEA = 64); r/z merged with gh later, i_n kept
  {
    const __bf16* x0 = x_bf + a_row0 * FEA;
    const __bf16* x1 = x_bf + a_row1 * FEA;
    const __bf16* wr = Wih + (0 * HID + n_g) * FEA;
    const __bf16* wz = Wih + (1 * HID + n_g) * FEA;
    const __bf16* wn = Wih + (2 * HID + n_g) * FEA;
#pragma unroll
    for (int k0 = 0; k0 < FEA; k0 += 32) {
      v16bf a0 = ld_a16(x0 + k0 + a_koff);
      v16bf a1 = ld_a16(x1 + k0 + a_koff);
      v16bf br = *(const v16bf*)(wr + k0 + b_koff);
      v16bf bz = *(const v16bf*)(wz + k0 + b_koff);
      v16bf bn = *(const v16bf*)(wn + k0 + b_koff);
      r0  = wmma_bf16(a0, br, r0);   r1  = wmma_bf16(a1, br, r1);
      z0  = wmma_bf16(a0, bz, z0);   z1  = wmma_bf16(a1, bz, z1);
      in0 = wmma_bf16(a0, bn, in0);  in1 = wmma_bf16(a1, bn, in1);
    }
  }
  // ---- gh = h @ Whh^T (K = HID = 512); h_n kept separate for r*h_n
  {
    const __bf16* h0 = h_in_bf + a_row0 * HID;
    const __bf16* h1 = h_in_bf + a_row1 * HID;
    const __bf16* wr = Whh + (0 * HID + n_g) * HID;
    const __bf16* wz = Whh + (1 * HID + n_g) * HID;
    const __bf16* wn = Whh + (2 * HID + n_g) * HID;
#pragma unroll 2
    for (int k0 = 0; k0 < HID; k0 += 32) {
      v16bf a0 = ld_a16(h0 + k0 + a_koff);
      v16bf a1 = ld_a16(h1 + k0 + a_koff);
      v16bf br = *(const v16bf*)(wr + k0 + b_koff);
      v16bf bz = *(const v16bf*)(wz + k0 + b_koff);
      v16bf bn = *(const v16bf*)(wn + k0 + b_koff);
      r0  = wmma_bf16(a0, br, r0);   r1  = wmma_bf16(a1, br, r1);
      z0  = wmma_bf16(a0, bz, z0);   z1  = wmma_bf16(a1, bz, z1);
      hn0 = wmma_bf16(a0, bn, hn0);  hn1 = wmma_bf16(a1, bn, hn1);
    }
  }

  // ---- fused gate epilogue (C/D layout: M = j + 8*lhi, N = lane15)
  const float brb = bih[n_g]           + bhh[n_g];
  const float bzb = bih[HID + n_g]     + bhh[HID + n_g];
  const float bin = bih[2 * HID + n_g];
  const float bhn = bhh[2 * HID + n_g];
  const int m_base0 = slab * 32 + lhi * 8;
#pragma unroll
  for (int j = 0; j < 8; ++j) {
    // first M tile
    {
      const int m = m_base0 + j;
      float r  = fast_sigmoid(r0[j] + brb);
      float z  = fast_sigmoid(z0[j] + bzb);
      float nn = fast_tanh(in0[j] + bin + r * (hn0[j] + bhn));
      float ho = h_in_f[m * HID + n_g];
      float hn = (1.0f - z) * nn + z * ho;
      h_out_f [m * HID + n_g] = hn;
      h_out_bf[m * HID + n_g] = (__bf16)hn;
    }
    // second M tile (+16 rows)
    {
      const int m = m_base0 + 16 + j;
      float r  = fast_sigmoid(r1[j] + brb);
      float z  = fast_sigmoid(z1[j] + bzb);
      float nn = fast_tanh(in1[j] + bin + r * (hn1[j] + bhn));
      float ho = h_in_f[m * HID + n_g];
      float hn = (1.0f - z) * nn + z * ho;
      h_out_f [m * HID + n_g] = hn;
      h_out_bf[m * HID + n_g] = (__bf16)hn;
    }
  }
}

// ---------------------------------------------------------------------------
// Decoder linear head: y = h @ lin_W^T + lin_b  ([512x512]@[512x64]).
// Grid: 16 blocks x 256 threads = 128 waves = 32x4 tiles.
// ---------------------------------------------------------------------------
__global__ __launch_bounds__(256) void lin_head_k(
    const __bf16* __restrict__ h_bf,    // [BAT, HID] bf16
    const __bf16* __restrict__ Wl,      // [FEA, HID] bf16
    const float*  __restrict__ bl,      // [FEA] fp32
    float*        __restrict__ y_out,   // [BAT, FEA] fp32 (d_out + t*B*F)
    __bf16*       __restrict__ x_next)  // [BAT, FEA] bf16 feedback
{
  const int wave   = (blockIdx.x * blockDim.x + threadIdx.x) >> 5;
  const int lane   = threadIdx.x & 31;
  const int tile_m = wave >> 2;  // 0..31
  const int tile_n = wave & 3;   // 0..3
  const int lane15 = lane & 15;
  const int lhi    = lane >> 4;

  const int a_row  = tile_m * 16 + lane15;
  const int a_koff = lhi * 16;
  const int b_koff = lhi * 16;
  const int f_g    = tile_n * 16 + lane15;

  v8f acc = {};
  const __bf16* hr = h_bf + a_row * HID;
  const __bf16* wr = Wl + f_g * HID;
#pragma unroll 4
  for (int k0 = 0; k0 < HID; k0 += 32) {
    v16bf a = ld_a16(hr + k0 + a_koff);
    acc = wmma_bf16(a, *(const v16bf*)(wr + k0 + b_koff), acc);
  }

  const float bias = bl[f_g];
  const int m_base = tile_m * 16 + lhi * 8;
#pragma unroll
  for (int j = 0; j < 8; ++j) {
    const int m = m_base + j;
    float y = acc[j] + bias;
    y_out [m * FEA + f_g] = y;
    x_next[m * FEA + f_g] = (__bf16)y;
  }
}

// ---------------------------------------------------------------------------
// fp32 -> bf16 conversion (weights + whole input tensor, once per call)
// ---------------------------------------------------------------------------
__global__ void cvt_f32_bf16_k(const float* __restrict__ s,
                               __bf16* __restrict__ d, int n) {
  int i = blockIdx.x * blockDim.x + threadIdx.x;
  const int stride = gridDim.x * blockDim.x;
  for (; i < n; i += stride) d[i] = (__bf16)s[i];
}

// ---------------------------------------------------------------------------
extern "C" void kernel_launch(void* const* d_in, const int* in_sizes, int n_in,
                              void* d_out, int out_size, void* d_ws, size_t ws_size,
                              hipStream_t stream) {
  (void)in_sizes; (void)n_in; (void)out_size; (void)ws_size;

  const float* inputs  = (const float*)d_in[0];   // [SEQ, BAT, FEA]
  const float* encWih  = (const float*)d_in[1];   // [3H, FEA]
  const float* encWhh  = (const float*)d_in[2];   // [3H, HID]
  const float* enc_bih = (const float*)d_in[3];
  const float* enc_bhh = (const float*)d_in[4];
  const float* decWih  = (const float*)d_in[5];
  const float* decWhh  = (const float*)d_in[6];
  const float* dec_bih = (const float*)d_in[7];
  const float* dec_bhh = (const float*)d_in[8];
  const float* linW    = (const float*)d_in[9];   // [FEA, HID]
  const float* linb    = (const float*)d_in[10];
  float* out = (float*)d_out;                     // [TOUT, BAT, FEA]

  // ---- workspace layout (256B aligned slices) ----
  char* w = (char*)d_ws;
  auto alloc = [&](size_t bytes) -> void* {
    void* p = (void*)w;
    w += (bytes + 255) & ~(size_t)255;
    return p;
  };
  __bf16* x_bf     = (__bf16*)alloc((size_t)SEQ * BAT * FEA * 2);
  __bf16* encWih_b = (__bf16*)alloc((size_t)3 * HID * FEA * 2);
  __bf16* encWhh_b = (__bf16*)alloc((size_t)3 * HID * HID * 2);
  __bf16* decWih_b = (__bf16*)alloc((size_t)3 * HID * FEA * 2);
  __bf16* decWhh_b = (__bf16*)alloc((size_t)3 * HID * HID * 2);
  __bf16* linW_b   = (__bf16*)alloc((size_t)FEA * HID * 2);
  float*  hf[2]; __bf16* hb[2];
  hf[0] = (float*)alloc((size_t)BAT * HID * 4);
  hf[1] = (float*)alloc((size_t)BAT * HID * 4);
  hb[0] = (__bf16*)alloc((size_t)BAT * HID * 2);
  hb[1] = (__bf16*)alloc((size_t)BAT * HID * 2);
  __bf16* xdec = (__bf16*)alloc((size_t)BAT * FEA * 2);

  // ---- one-time (per call) conversions to bf16 ----
  auto cvt = [&](const float* s, __bf16* d, int n) {
    int blocks = (n + 256 * 8 - 1) / (256 * 8);
    if (blocks > 2048) blocks = 2048;
    cvt_f32_bf16_k<<<blocks, 256, 0, stream>>>(s, d, n);
  };
  cvt(inputs, x_bf, SEQ * BAT * FEA);
  cvt(encWih, encWih_b, 3 * HID * FEA);
  cvt(encWhh, encWhh_b, 3 * HID * HID);
  cvt(decWih, decWih_b, 3 * HID * FEA);
  cvt(decWhh, decWhh_b, 3 * HID * HID);
  cvt(linW,   linW_b,   FEA * HID);

  // ---- initial state: h0 = 0, decoder x0 = 0 ----
  hipMemsetAsync(hf[0], 0, (size_t)BAT * HID * 4, stream);
  hipMemsetAsync(hb[0], 0, (size_t)BAT * HID * 2, stream);
  hipMemsetAsync(xdec,  0, (size_t)BAT * FEA * 2, stream);

  int cur = 0;

  // ---- encoder: 336 sequential fused GRU-cell steps ----
  for (int t = 0; t < SEQ; ++t) {
    const int nxt = cur ^ 1;
    gru_cell_k<<<64, 256, 0, stream>>>(
        x_bf + (size_t)t * BAT * FEA,
        encWih_b, encWhh_b, enc_bih, enc_bhh,
        hb[cur], hf[cur], hf[nxt], hb[nxt]);
    cur = nxt;
  }

  // ---- decoder: 48 autoregressive steps (GRU cell + linear head) ----
  for (int t = 0; t < TOUT; ++t) {
    const int nxt = cur ^ 1;
    gru_cell_k<<<64, 256, 0, stream>>>(
        xdec,
        decWih_b, decWhh_b, dec_bih, dec_bhh,
        hb[cur], hf[cur], hf[nxt], hb[nxt]);
    lin_head_k<<<16, 256, 0, stream>>>(
        hb[nxt], linW_b, linb,
        out + (size_t)t * BAT * FEA, xdec);
    cur = nxt;
  }
}
